// MambaBlockTest_79182017069456
// MI455X (gfx1250) — matrix operand order repaired and checked by reading, not compiled, not run on hardware
//
#include <hip/hip_runtime.h>
#include <hip/hip_bf16.h>

// ---------------------------------------------------------------------------
// Mamba block forward, MI455X (gfx1250, wave32, WMMA + async-to-LDS).
//   0) pack: x -> bf16, W_in/W_xproj -> bf16 transposed (K-contiguous B frags)
//   1) xz = x @ W_in            (WMMA bf16 16x16x32, A staged in LDS async)
//   2) u_c = silu(conv(u)+b)    (elementwise, also emits bf16 copy)
//   3) x_dbl = u_c @ W_xproj    (WMMA bf16, same kernel)
//   4) fused scan: dt-proj+softplus+dA+dBu+recurrence+C-contract+skip+gate+mean
//   5) out = g @ W_out @ W_cls + b_cls   (mean commuted through linear maps)
// ---------------------------------------------------------------------------

#define D_MODEL 256
#define D_INNER 512
#define D_STATE 16
#define KCONV   4
#define SEQ_LEN 2048
#define N_CLS   10
#define DT_RANK 16
#define BATCH   4
#define BL      (BATCH * SEQ_LEN)          // 8192 rows
#define D_XZ    (2 * D_INNER)              // 1024
#define D_XDBL  (DT_RANK + 2 * D_STATE)    // 48
#define MAX_K   512                        // largest GEMM K (u_c @ W_xproj)

typedef __attribute__((ext_vector_type(16))) __bf16 v16bf;
typedef __attribute__((ext_vector_type(8)))  __bf16 bf16x8;
typedef __attribute__((ext_vector_type(8)))  float  v8f;

// ---------------------------------------------------------------------------
// Elementwise f32 -> bf16 pack.
// ---------------------------------------------------------------------------
__global__ void mamba_pack_bf16(const float* __restrict__ in,
                                __bf16* __restrict__ out, int n) {
    const int i = blockIdx.x * blockDim.x + threadIdx.x;
    if (i < n) out[i] = (__bf16)in[i];
}

// ---------------------------------------------------------------------------
// Transposing f32(KxN) -> bf16(NxK) pack, so B fragments are K-contiguous.
// ---------------------------------------------------------------------------
__global__ void mamba_packT_bf16(const float* __restrict__ in,
                                 __bf16* __restrict__ out, int K, int N) {
    const int i = blockIdx.x * blockDim.x + threadIdx.x;
    if (i < K * N) {
        const int k = i / N, n = i - k * N;
        out[(size_t)n * K + k] = (__bf16)in[i];
    }
}

// ---------------------------------------------------------------------------
// WMMA GEMM: C(MxN) = A(MxK) @ B(KxN); A bf16 row-major, BT = B^T bf16 (NxK),
// C f32 row-major. One wave per 16x16 C tile; the block's 16xK A tile (a
// contiguous byte range, since we take 16 full rows) is staged into LDS with
// gfx1250 async loads and shared by all waves (each wave = one N tile).
// grid = (M/16, N/(16*wavesPerBlock)), block = 32*wavesPerBlock.
// ---------------------------------------------------------------------------
__global__ void mamba_wmma_gemm_bf16(const __bf16* __restrict__ A,
                                     const __bf16* __restrict__ BT,
                                     float* __restrict__ C,
                                     int M, int K, int N) {
    __shared__ __bf16 tileA[16 * MAX_K];                 // 16 KB
    const int tid  = threadIdx.x;
    const int lane = tid & 31;
    const int wave = tid >> 5;
    const int wpb  = blockDim.x >> 5;
    const int ln   = lane & 15;          // M row (A/C) / N col (B/C) in tile
    const int g    = lane >> 4;          // lane-half selector
    const int mt   = blockIdx.x;
    const int nt   = blockIdx.y * wpb + wave;

    // ---- async-stage 16 full A rows (contiguous 16*K*2 bytes) into LDS ----
    {
        const char* src   = (const char*)(A + (size_t)mt * 16 * K);
        const int   bytes = 16 * K * 2;
        for (int off = tid * 16; off < bytes; off += blockDim.x * 16) {
            unsigned    la = (unsigned)(uintptr_t)((char*)tileA + off);
            const char* ga = src + off;
            asm volatile("global_load_async_to_lds_b128 %0, %1, off"
                         :: "v"(la), "v"(ga) : "memory");
        }
        asm volatile("s_wait_asynccnt 0x0" ::: "memory");
    }
    __syncthreads();

    const __bf16* arow = tileA + ln * K;                       // LDS
    const __bf16* brow = BT + (size_t)(nt * 16 + ln) * K;      // global

    // Fragment layouts (ISA 7.12.2, wave32):
    //  A 16x32: elem e<8  -> K = kc + 8g + e       (one b128 run)
    //           elem e>=8 -> K = kc + 16 + 8g + (e-8)
    //  B 32x16: same K mapping as A                 (two b128 runs from B^T)
    v8f c = {};
#pragma unroll 2
    for (int kc = 0; kc < K; kc += 32) {
        if (kc + 32 < K)
            __builtin_prefetch(brow + kc + 32, 0, 1);          // global_prefetch
        bf16x8 alo = *(const bf16x8*)(arow + kc + 8 * g);      // ds_load_b128
        bf16x8 ahi = *(const bf16x8*)(arow + kc + 16 + 8 * g);
        bf16x8 blo = *(const bf16x8*)(brow + kc + 8 * g);      // global_load_b128
        bf16x8 bhi = *(const bf16x8*)(brow + kc + 16 + 8 * g);
        v16bf a = __builtin_shufflevector(alo, ahi, 0, 1, 2, 3, 4, 5, 6, 7,
                                          8, 9, 10, 11, 12, 13, 14, 15);
        v16bf b = __builtin_shufflevector(blo, bhi, 0, 1, 2, 3, 4, 5, 6, 7,
                                          8, 9, 10, 11, 12, 13, 14, 15);
        c = __builtin_amdgcn_wmma_f32_16x16x32_bf16(
                false, a, false, b, (short)0, c, false, false);
    }
#pragma unroll
    for (int r = 0; r < 8; ++r) {
        const int row = mt * 16 + r + 8 * g;                   // C: M = r + 8g
        C[(size_t)row * N + nt * 16 + ln] = c[r];
    }
}

// ---------------------------------------------------------------------------
// Depthwise causal conv (K=4) + bias + SiLU; u lives in xz[:, 0:512].
// Emits both f32 (for the scan) and bf16 (for the x_proj WMMA GEMM).
// ---------------------------------------------------------------------------
__global__ void mamba_conv_silu(const float* __restrict__ xz,
                                const float* __restrict__ Wc,
                                const float* __restrict__ bc,
                                float* __restrict__ uc,
                                __bf16* __restrict__ ucbf) {
    const int idx = blockIdx.x * blockDim.x + threadIdx.x;    // BL*D_INNER
    const int d   = idx & (D_INNER - 1);
    const int bl  = idx >> 9;
    const int l   = bl & (SEQ_LEN - 1);
    const int b0  = bl - l;                                   // b*SEQ_LEN

    float acc = bc[d];
#pragma unroll
    for (int k = 0; k < KCONV; ++k) {
        const int p = l - (KCONV - 1) + k;
        if (p >= 0)
            acc += xz[(size_t)(b0 + p) * D_XZ + d] * Wc[d * KCONV + k];
    }
    acc = acc / (1.0f + __expf(-acc));                        // silu
    uc[idx]   = acc;
    ucbf[idx] = (__bf16)acc;
}

// ---------------------------------------------------------------------------
// Fused selective scan. One thread per (b, d): 2048 independent scans, state
// h[16] in registers. Fuses dt-projection (K=16), softplus, dA=exp(dt*A),
// dBu, recurrence, <h,C>, +u*D skip, *silu(z) gate and the mean over L.
// Never materializes dA/dBu/y — writes only g (4x512).
// ---------------------------------------------------------------------------
__global__ void mamba_scan(const float* __restrict__ xdbl,
                           const float* __restrict__ uc,
                           const float* __restrict__ xz,
                           const float* __restrict__ W_dt,
                           const float* __restrict__ b_dt,
                           const float* __restrict__ A_log,
                           const float* __restrict__ Dp,
                           float* __restrict__ g) {
    const int t = blockIdx.x * blockDim.x + threadIdx.x;      // 0 .. B*D_INNER
    const int d = t & (D_INNER - 1);
    const int b = t >> 9;

    float An[D_STATE], wdt[DT_RANK], h[D_STATE];
#pragma unroll
    for (int n = 0; n < D_STATE; ++n) {
        An[n] = -__expf(A_log[d * D_STATE + n]);
        h[n]  = 0.0f;
    }
#pragma unroll
    for (int r = 0; r < DT_RANK; ++r)
        wdt[r] = W_dt[r * D_INNER + d];                       // coalesced in d

    const float bdt = b_dt[d];
    const float Dd  = Dp[d];
    float acc = 0.0f;

    for (int l = 0; l < SEQ_LEN; ++l) {
        const size_t row = (size_t)b * SEQ_LEN + l;
        const float* xr  = xdbl + row * D_XDBL;               // wave-uniform

        float dtv = bdt;
#pragma unroll
        for (int r = 0; r < DT_RANK; ++r)
            dtv += xr[r] * wdt[r];
        const float dt = (dtv > 20.0f) ? dtv : log1pf(__expf(dtv)); // softplus

        const float u  = uc[row * D_INNER + d];               // coalesced
        const float du = dt * u;
        float y = 0.0f;
#pragma unroll
        for (int n = 0; n < D_STATE; ++n) {
            const float dA = __expf(dt * An[n]);
            h[n] = dA * h[n] + du * xr[DT_RANK + n];          // Bp
            y   += h[n] * xr[DT_RANK + D_STATE + n];          // Cp
        }
        y += u * Dd;
        const float z = xz[row * D_XZ + D_INNER + d];         // z half
        acc += y * (z / (1.0f + __expf(-z)));                 // * silu(z)
    }
    g[t] = acc * (1.0f / (float)SEQ_LEN);
}

// ---------------------------------------------------------------------------
// Head: out[b,c] = (g[b,:] @ W_out) @ W_cls + b_cls.  One 256-thread block.
// ---------------------------------------------------------------------------
__global__ void mamba_head(const float* __restrict__ g,
                           const float* __restrict__ W_out,
                           const float* __restrict__ W_cls,
                           const float* __restrict__ b_cls,
                           float* __restrict__ out) {
    __shared__ float tmp[BATCH * D_MODEL];
    const int m = threadIdx.x;                                // 0..255
    for (int b = 0; b < BATCH; ++b) {
        float s = 0.0f;
        for (int d = 0; d < D_INNER; ++d)
            s += g[b * D_INNER + d] * W_out[(size_t)d * D_MODEL + m];
        tmp[b * D_MODEL + m] = s;
    }
    __syncthreads();
    if (m < BATCH * N_CLS) {
        const int b = m / N_CLS, c = m % N_CLS;
        float o = b_cls[c];
        for (int mm = 0; mm < D_MODEL; ++mm)
            o += tmp[b * D_MODEL + mm] * W_cls[mm * N_CLS + c];
        out[b * N_CLS + c] = o;
    }
}

// ---------------------------------------------------------------------------
extern "C" void kernel_launch(void* const* d_in, const int* in_sizes, int n_in,
                              void* d_out, int out_size, void* d_ws, size_t ws_size,
                              hipStream_t stream) {
    (void)in_sizes; (void)n_in; (void)out_size; (void)ws_size;

    const float* x      = (const float*)d_in[0];   // (4,2048,256)
    const float* W_in   = (const float*)d_in[1];   // (256,1024)
    const float* W_conv = (const float*)d_in[2];   // (512,4)
    const float* b_conv = (const float*)d_in[3];   // (512)
    const float* W_xprj = (const float*)d_in[4];   // (512,48)
    const float* W_dt   = (const float*)d_in[5];   // (16,512)
    const float* b_dt   = (const float*)d_in[6];   // (512)
    const float* A_log  = (const float*)d_in[7];   // (512,16)
    const float* Dp     = (const float*)d_in[8];   // (512)
    const float* W_out  = (const float*)d_in[9];   // (512,256)
    const float* W_cls  = (const float*)d_in[10];  // (256,10)
    const float* b_cls  = (const float*)d_in[11];  // (10)
    float* out = (float*)d_out;                    // (4,10)

    // workspace layout: f32 region first, then bf16 region
    float* ws    = (float*)d_ws;
    float* xz    = ws;                                   // 8192*1024 f32
    float* u_c   = xz    + (size_t)BL * D_XZ;            // 8192*512  f32
    float* x_dbl = u_c   + (size_t)BL * D_INNER;         // 8192*48   f32
    float* gbuf  = x_dbl + (size_t)BL * D_XDBL;          // 2048      f32
    __bf16* xbf  = (__bf16*)(gbuf + BATCH * D_INNER);    // 8192*256  bf16
    __bf16* winT = xbf  + (size_t)BL * D_MODEL;          // 1024*256  bf16
    __bf16* ucbf = winT + (size_t)D_XZ * D_MODEL;        // 8192*512  bf16
    __bf16* wxT  = ucbf + (size_t)BL * D_INNER;          // 48*512    bf16

    // 0) pack operands to bf16 (weights transposed -> K-contiguous B frags)
    mamba_pack_bf16<<<(BL * D_MODEL) / 256, 256, 0, stream>>>(x, xbf, BL * D_MODEL);
    mamba_packT_bf16<<<(D_MODEL * D_XZ) / 256, 256, 0, stream>>>(W_in, winT, D_MODEL, D_XZ);
    mamba_packT_bf16<<<(D_INNER * D_XDBL + 255) / 256, 256, 0, stream>>>(W_xprj, wxT, D_INNER, D_XDBL);

    // 1) xz = x @ W_in   (8192x256 @ 256x1024): 8 waves/block, 8 N-tiles
    mamba_wmma_gemm_bf16<<<dim3(BL / 16, D_XZ / (16 * 8)), 256, 0, stream>>>(
        xbf, winT, xz, BL, D_MODEL, D_XZ);

    // 2) u_c = silu(conv(u) + b_conv)  (+ bf16 copy)
    mamba_conv_silu<<<(BL * D_INNER) / 256, 256, 0, stream>>>(
        xz, W_conv, b_conv, u_c, ucbf);

    // 3) x_dbl = u_c @ W_xproj  (8192x512 @ 512x48): 3 waves/block, 3 N-tiles
    mamba_wmma_gemm_bf16<<<dim3(BL / 16, 1), 96, 0, stream>>>(
        ucbf, wxT, x_dbl, BL, D_INNER, D_XDBL);

    // 4) fused selective scan -> gated mean g (4x512)
    mamba_scan<<<(BATCH * D_INNER) / 256, 256, 0, stream>>>(
        x_dbl, u_c, xz, W_dt, b_dt, A_log, Dp, gbuf);

    // 5) head: g @ W_out @ W_cls + b_cls
    mamba_head<<<1, D_MODEL, 0, stream>>>(gbuf, W_out, W_cls, b_cls, out);
}
